// STGACN_84138409328925
// MI455X (gfx1250) — compile-verified, matrix operand order checked by reading, not run on hardware
//
#include <hip/hip_runtime.h>
#include <hip/hip_bf16.h>
#include <hip/hip_fp16.h>

// ---------------------------------------------------------------------------
// ST-GCN (GATv2) forward for MI455X / gfx1250.
//  - all dense matmuls (GAT projections, im2col'd temporal convs, FC1, FC2)
//    -> v_wmma_f32_16x16x32_f16, one 16x16 tile per wave32
//  - conv A-tiles staged global->LDS with global_load_async_to_lds_b128
//    (ASYNCcnt) then read back as ds_load_b128 fragments
//  - sparse softmax aggregation -> f32 atomics (entire state fits in 192MB L2)
// ---------------------------------------------------------------------------

typedef __attribute__((ext_vector_type(16))) _Float16 v16h;
typedef __attribute__((ext_vector_type(8)))  _Float16 v8h;
typedef __attribute__((ext_vector_type(8)))  float    v8f;

#define BB    128
#define TT    900
#define NND   115200            // B*T nodes
#define ECNT  1382400           // 12*N edges
#define ENTOT 1497600           // edges + self loops
#define EPS_  1e-5f

// ---- fragment loader: 16-bit A/B layout (ISA 7.12.2) ----------------------
// lane (0..15 | 16..31) holds row/col (lane&15); K indices per 16-half frag:
//   elems 0..7  -> K = kk +  0 + hi*8 + (0..7)
//   elems 8..15 -> K = kk + 16 + hi*8 + (0..7)
// i.e. two contiguous 16B loads from a K-major row (global or LDS).
static __device__ __forceinline__ v16h ld_frag(const _Float16* p, int hi) {
    v16h r;
    ((v8h*)&r)[0] = *(const v8h*)(p + hi * 8);
    ((v8h*)&r)[1] = *(const v8h*)(p + 16 + hi * 8);
    return r;
}

static __device__ __forceinline__ uint32_t lds_offset_of(const void* p) {
    return (uint32_t)(size_t)(__attribute__((address_space(3))) const void*)p;
}

// ---------------------------------------------------------------------------
// generic helpers
// ---------------------------------------------------------------------------
__global__ void zero_f32_k(float* __restrict__ p, size_t n) {
    size_t i = (size_t)blockIdx.x * blockDim.x + threadIdx.x;
    if (i < n) p[i] = 0.0f;
}

__global__ void f32_to_f16_k(const float* __restrict__ in, _Float16* __restrict__ out, size_t n) {
    size_t i = (size_t)blockIdx.x * blockDim.x + threadIdx.x;
    if (i < n) out[i] = (_Float16)in[i];
}

// ---------------------------------------------------------------------------
// first temporal conv (CIN=3, K=15: too small for WMMA) + ReLU, [B,C,T]
// ---------------------------------------------------------------------------
template<int CIN, int COUT>
__global__ void conv1d_relu_k(const float* __restrict__ x, const float* __restrict__ w,
                              const float* __restrict__ bias, float* __restrict__ y) {
    const int idx = blockIdx.x * blockDim.x + threadIdx.x;
    constexpr int TOT = BB * COUT * TT;
    if (idx >= TOT) return;
    const int t  = idx % TT;
    const int co = (idx / TT) % COUT;
    const int b  = idx / (TT * COUT);
    const float* xb = x + (size_t)b * CIN * TT;
    const float* wc = w + (size_t)co * CIN * 5;
    float acc = bias[co];
    for (int ci = 0; ci < CIN; ++ci) {
        const float* xrow = xb + ci * TT;
        const float* wrow = wc + ci * 5;
#pragma unroll
        for (int k = 0; k < 5; ++k) {
            const int tt = t + k - 2;
            const float xv = (tt >= 0 && tt < TT) ? xrow[tt] : 0.0f;
            acc = fmaf(wrow[k], xv, acc);
        }
    }
    y[idx] = fmaxf(acc, 0.0f);
}

// ---------------------------------------------------------------------------
// im2col for K=5 'same' conv: IM[(b*T+t), ci*5+k] = xpad[b, ci, t+k-2] (f16)
// ---------------------------------------------------------------------------
template<int CIN>
__global__ void im2col_f16_k(const float* __restrict__ x, _Float16* __restrict__ im) {
    constexpr int K = CIN * 5;
    const size_t i = (size_t)blockIdx.x * blockDim.x + threadIdx.x;
    if (i >= (size_t)NND * K) return;
    const int kcol = (int)(i % K);
    const size_t m = i / K;
    const int b = (int)(m / TT), t = (int)(m % TT);
    const int ci = kcol / 5, k = kcol % 5;
    const int tt = t + k - 2;
    const float v = (tt >= 0 && tt < TT) ? x[((size_t)b * CIN + ci) * TT + tt] : 0.0f;
    im[i] = (_Float16)v;
}

// ---------------------------------------------------------------------------
// conv-as-GEMM with WMMA. A tile (16 rows x K, contiguous in IM) is staged
// into LDS via async loads (ASYNCcnt) with a padded pitch (bank-conflict
// free ds_load_b128), B = conv weight [CO, CIN*5] f16 straight from global.
// Result scattered back to [B, COUT, T] with bias+ReLU.
// ---------------------------------------------------------------------------
template<int K, int COUT>
__global__ __launch_bounds__(32) void conv_wmma_k(
    const _Float16* __restrict__ IM, const _Float16* __restrict__ w16,
    const float* __restrict__ bias, float* __restrict__ out) {
    constexpr int KP = K + 8;                         // LDS row pitch (halves)
    __shared__ _Float16 smA[16 * KP];
    const int lane = threadIdx.x & 31;
    const int hi = lane >> 4, m = lane & 15;
    constexpr int CPR = (K * 2) / 16;                 // 16B chunks per row

    // ---- stage 16xK A tile: global (linear) -> LDS (pitched) ----
    const _Float16* gA = IM + (size_t)blockIdx.x * 16 * K;
#if defined(__gfx1250__)
    const uint32_t lbase = lds_offset_of(smA);
#pragma unroll
    for (int j = 0; j < (16 * K) / (8 * 32); ++j) {
        const int c = lane + 32 * j;
        const int row = c / CPR, within = c % CPR;
        const uint64_t g = (uint64_t)(size_t)(gA + (size_t)c * 8);
        asm volatile("global_load_async_to_lds_b128 %0, %1, off"
                     :: "v"(lbase + (uint32_t)(row * KP * 2 + within * 16)), "v"(g)
                     : "memory");
    }
    asm volatile("s_wait_asynccnt 0" ::: "memory");
#else
    for (int j = 0; j < (16 * K) / (8 * 32); ++j) {
        const int c = lane + 32 * j;
        const int row = c / CPR, within = c % CPR;
        *(v8h*)(smA + row * KP + within * 8) = *(const v8h*)(gA + (size_t)c * 8);
    }
    __syncthreads();
#endif

    const size_t ncol = (size_t)blockIdx.y * 16 + m;
    const _Float16* pa = smA + (size_t)m * KP;        // LDS -> ds_load_b128 frags
    const _Float16* pw = w16 + ncol * K;
    v8f acc = {};
#pragma unroll
    for (int kk = 0; kk < K; kk += 32) {
        v16h a = ld_frag(pa + kk, hi);
        v16h b = ld_frag(pw + kk, hi);
        acc = __builtin_amdgcn_wmma_f32_16x16x32_f16(false, a, false, b, (short)0, acc, false, false);
    }
    const int ocol = (int)blockIdx.y * 16 + m;
    const float bv = bias[ocol];
#pragma unroll
    for (int r = 0; r < 8; ++r) {
        const int mrow = (int)blockIdx.x * 16 + r + hi * 8;
        const int b = mrow / TT, t = mrow - b * TT;
        out[((size_t)b * COUT + ocol) * TT + t] = fmaxf(acc[r] + bv, 0.0f);
    }
}

// ---------------------------------------------------------------------------
// GATv2 projections: XL = H @ WL^T + bl ; XR = H @ WR^T + br   (WMMA f16)
// ---------------------------------------------------------------------------
template<int C>
__global__ __launch_bounds__(32) void gat_linear_wmma_k(
    const _Float16* __restrict__ h16,
    const _Float16* __restrict__ wl16, const float* __restrict__ bl,
    const _Float16* __restrict__ wr16, const float* __restrict__ br,
    float* __restrict__ xl, float* __restrict__ xr) {
    const int lane = threadIdx.x & 31;
    const int hi   = lane >> 4;
    const int m    = lane & 15;
    const size_t row  = (size_t)blockIdx.x * 16 + m;
    const size_t ncol = (size_t)blockIdx.y * 16 + m;
    v8f accL = {}; v8f accR = {};
    const _Float16* pa = h16  + row  * C;
    const _Float16* pl = wl16 + ncol * C;
    const _Float16* pr = wr16 + ncol * C;
#pragma unroll
    for (int kk = 0; kk < C; kk += 32) {
        v16h a  = ld_frag(pa + kk, hi);
        v16h bL = ld_frag(pl + kk, hi);
        v16h bR = ld_frag(pr + kk, hi);
        accL = __builtin_amdgcn_wmma_f32_16x16x32_f16(false, a, false, bL, (short)0, accL, false, false);
        accR = __builtin_amdgcn_wmma_f32_16x16x32_f16(false, a, false, bR, (short)0, accR, false, false);
    }
    const int ocol = (int)blockIdx.y * 16 + m;
    const float bvl = bl[ocol], bvr = br[ocol];
#pragma unroll
    for (int r = 0; r < 8; ++r) {
        const size_t orow = (size_t)blockIdx.x * 16 + r + hi * 8;
        xl[orow * C + ocol] = accL[r] + bvl;
        xr[orow * C + ocol] = accR[r] + bvr;
    }
}

// ---------------------------------------------------------------------------
// edge pass 1: e = att . leaky_relu(xl[src]+xr[dst]); ex = exp(e) (softmax is
// shift-invariant -> segment_max elided); segment-sum via L2 atomics.
// ---------------------------------------------------------------------------
template<int C>
__global__ void edge_scores_k(const int* __restrict__ eidx,
                              const float* __restrict__ xl, const float* __restrict__ xr,
                              const float* __restrict__ att,
                              float* __restrict__ exbuf, float* __restrict__ ssum) {
    const int idx = blockIdx.x * blockDim.x + threadIdx.x;
    if (idx >= ENTOT) return;
    const int s = (idx < ECNT) ? eidx[idx] : (idx - ECNT);
    const int d = (idx < ECNT) ? eidx[ECNT + idx] : (idx - ECNT);
    const float4* pa = (const float4*)(xl + (size_t)s * C);
    const float4* pb = (const float4*)(xr + (size_t)d * C);
    const float4* pt = (const float4*)att;
    float e = 0.0f;
#pragma unroll
    for (int q = 0; q < C / 4; ++q) {
        const float4 a = pa[q], b = pb[q], t = pt[q];
        float v0 = a.x + b.x; v0 = (v0 > 0.f) ? v0 : 0.2f * v0;
        float v1 = a.y + b.y; v1 = (v1 > 0.f) ? v1 : 0.2f * v1;
        float v2 = a.z + b.z; v2 = (v2 > 0.f) ? v2 : 0.2f * v2;
        float v3 = a.w + b.w; v3 = (v3 > 0.f) ? v3 : 0.2f * v3;
        e = fmaf(t.x, v0, fmaf(t.y, v1, fmaf(t.z, v2, fmaf(t.w, v3, e))));
    }
    const float ex = __expf(e);
    exbuf[idx] = ex;
    atomicAdd(&ssum[d], ex);
}

// edge pass 2: out[dst] += (ex/sum[dst]) * xl[src]
template<int C>
__global__ void edge_aggregate_k(const int* __restrict__ eidx,
                                 const float* __restrict__ xl,
                                 const float* __restrict__ exbuf,
                                 const float* __restrict__ ssum,
                                 float* __restrict__ gout) {
    const int idx = blockIdx.x * blockDim.x + threadIdx.x;
    if (idx >= ENTOT) return;
    const int s = (idx < ECNT) ? eidx[idx] : (idx - ECNT);
    const int d = (idx < ECNT) ? eidx[ECNT + idx] : (idx - ECNT);
    const float alpha = exbuf[idx] / ssum[d];
    const float4* pa = (const float4*)(xl + (size_t)s * C);
    float* pd = gout + (size_t)d * C;
#pragma unroll
    for (int q = 0; q < C / 4; ++q) {
        const float4 a = pa[q];
        atomicAdd(&pd[q * 4 + 0], alpha * a.x);
        atomicAdd(&pd[q * 4 + 1], alpha * a.y);
        atomicAdd(&pd[q * 4 + 2], alpha * a.z);
        atomicAdd(&pd[q * 4 + 3], alpha * a.w);
    }
}

// h = relu(h + gbias) in place, accumulate per-channel sum / sumsq
template<int C>
__global__ void gat_post_stats_k(float* __restrict__ h, const float* __restrict__ gbias,
                                 float* __restrict__ stats) {
    __shared__ float sm[2 * C];
    const int tid = threadIdx.x;
    for (int i = tid; i < 2 * C; i += blockDim.x) sm[i] = 0.0f;
    __syncthreads();
    const size_t i = (size_t)blockIdx.x * blockDim.x + tid;
    if (i < (size_t)NND * C) {
        const int c = (int)(i % C);
        float v = h[i] + gbias[c];
        v = fmaxf(v, 0.0f);
        h[i] = v;
        atomicAdd(&sm[c], v);
        atomicAdd(&sm[C + c], v * v);
    }
    __syncthreads();
    for (int i2 = tid; i2 < 2 * C; i2 += blockDim.x) atomicAdd(&stats[i2], sm[i2]);
}

template<int C>
__global__ void bn_apply_k(float* __restrict__ h, const float* __restrict__ stats,
                           const float* __restrict__ g, const float* __restrict__ b) {
    const size_t i = (size_t)blockIdx.x * blockDim.x + threadIdx.x;
    if (i >= (size_t)NND * C) return;
    const int c = (int)(i % C);
    const float inv  = 1.0f / (float)NND;
    const float mean = stats[c] * inv;
    const float var  = stats[C + c] * inv - mean * mean;
    h[i] = (h[i] - mean) * rsqrtf(var + EPS_) * g[c] + b[c];
}

// ---------------------------------------------------------------------------
// FC1: [128,57600] @ [128,57600]^T, split-K WMMA with f32 atomic reduction
// ---------------------------------------------------------------------------
__global__ __launch_bounds__(32) void fc1_wmma_splitk_k(
    const _Float16* __restrict__ X, const _Float16* __restrict__ W, float* __restrict__ out) {
    constexpr int K = 57600, SLICE = 7200;    // 8 slices x 225 wmma
    const int lane = threadIdx.x & 31;
    const int hi = lane >> 4, m = lane & 15;
    const size_t row = (size_t)blockIdx.x * 16 + m;
    const size_t nc  = (size_t)blockIdx.y * 16 + m;
    const int k0 = blockIdx.z * SLICE;
    v8f acc = {};
    const _Float16* px = X + row * K;
    const _Float16* pw = W + nc * K;
    for (int kk = k0; kk < k0 + SLICE; kk += 32) {
        v16h a = ld_frag(px + kk, hi);
        v16h b = ld_frag(pw + kk, hi);
        acc = __builtin_amdgcn_wmma_f32_16x16x32_f16(false, a, false, b, (short)0, acc, false, false);
    }
    const int ocol = (int)blockIdx.y * 16 + m;
#pragma unroll
    for (int r = 0; r < 8; ++r)
        atomicAdd(&out[((size_t)blockIdx.x * 16 + r + hi * 8) * 128 + ocol], acc[r]);
}

__global__ void bias_relu_k(float* __restrict__ h, const float* __restrict__ b, int rows, int cols) {
    const int i = blockIdx.x * blockDim.x + threadIdx.x;
    if (i >= rows * cols) return;
    h[i] = fmaxf(h[i] + b[i % cols], 0.0f);
}

// FC2: [128,128] @ [64,128]^T + bias, ReLU (single-pass WMMA)
__global__ __launch_bounds__(32) void fc2_wmma_k(
    const _Float16* __restrict__ X, const _Float16* __restrict__ W,
    const float* __restrict__ bias, float* __restrict__ out) {
    constexpr int K = 128;
    const int lane = threadIdx.x & 31;
    const int hi = lane >> 4, m = lane & 15;
    const size_t row = (size_t)blockIdx.x * 16 + m;
    const size_t nc  = (size_t)blockIdx.y * 16 + m;
    v8f acc = {};
    const _Float16* px = X + row * K;
    const _Float16* pw = W + nc * K;
#pragma unroll
    for (int kk = 0; kk < K; kk += 32) {
        v16h a = ld_frag(px + kk, hi);
        v16h b = ld_frag(pw + kk, hi);
        acc = __builtin_amdgcn_wmma_f32_16x16x32_f16(false, a, false, b, (short)0, acc, false, false);
    }
    const int ocol = (int)blockIdx.y * 16 + m;
    const float bv = bias[ocol];
#pragma unroll
    for (int r = 0; r < 8; ++r)
        out[((size_t)blockIdx.x * 16 + r + hi * 8) * 64 + ocol] = fmaxf(acc[r] + bv, 0.0f);
}

// per-feature batchnorm over a small [rows, cols] activation (one thread/col)
__global__ void head_bn_k(float* __restrict__ h, const float* __restrict__ g,
                          const float* __restrict__ b, int rows, int cols) {
    const int c = blockIdx.x * blockDim.x + threadIdx.x;
    if (c >= cols) return;
    float s = 0.0f, s2 = 0.0f;
    for (int r = 0; r < rows; ++r) { const float v = h[r * cols + c]; s += v; s2 = fmaf(v, v, s2); }
    const float mean = s / rows;
    const float var  = s2 / rows - mean * mean;
    const float sc = rsqrtf(var + EPS_) * g[c];
    const float sh = b[c] - mean * sc;
    for (int r = 0; r < rows; ++r) h[r * cols + c] = fmaf(h[r * cols + c], sc, sh);
}

// FC3 (3 classes) + sigmoid; writes [sigmoid | logits] into d_out
__global__ void fc3_out_k(const float* __restrict__ h, const float* __restrict__ w,
                          const float* __restrict__ b, float* __restrict__ out) {
    const int idx = blockIdx.x * blockDim.x + threadIdx.x;
    if (idx >= BB * 3) return;
    const int n = idx % 3, bb = idx / 3;
    float acc = b[n];
    for (int k = 0; k < 64; ++k) acc = fmaf(h[bb * 64 + k], w[n * 64 + k], acc);
    out[BB * 3 + idx] = acc;
    out[idx] = 1.0f / (1.0f + __expf(-acc));
}

// ---------------------------------------------------------------------------
// launcher
// ---------------------------------------------------------------------------
static inline unsigned gblk(size_t n, unsigned t) { return (unsigned)((n + t - 1) / t); }

extern "C" void kernel_launch(void* const* d_in, const int* in_sizes, int n_in,
                              void* d_out, int out_size, void* d_ws, size_t ws_size,
                              hipStream_t stream) {
    (void)in_sizes; (void)n_in; (void)out_size; (void)ws_size;

    const float* x      = (const float*)d_in[0];
    const int*   eidx   = (const int*)d_in[1];
    const float* b1_tc1_w = (const float*)d_in[2];  const float* b1_tc1_b = (const float*)d_in[3];
    const float* b1_wl  = (const float*)d_in[4];    const float* b1_bl  = (const float*)d_in[5];
    const float* b1_wr  = (const float*)d_in[6];    const float* b1_br  = (const float*)d_in[7];
    const float* b1_att = (const float*)d_in[8];    const float* b1_gbias = (const float*)d_in[9];
    const float* b1_bn_g = (const float*)d_in[10];  const float* b1_bn_b = (const float*)d_in[11];
    const float* b1_tc2_w = (const float*)d_in[12]; const float* b1_tc2_b = (const float*)d_in[13];
    const float* b2_tc1_w = (const float*)d_in[14]; const float* b2_tc1_b = (const float*)d_in[15];
    const float* b2_wl  = (const float*)d_in[16];   const float* b2_bl  = (const float*)d_in[17];
    const float* b2_wr  = (const float*)d_in[18];   const float* b2_br  = (const float*)d_in[19];
    const float* b2_att = (const float*)d_in[20];   const float* b2_gbias = (const float*)d_in[21];
    const float* b2_bn_g = (const float*)d_in[22];  const float* b2_bn_b = (const float*)d_in[23];
    const float* b2_tc2_w = (const float*)d_in[24]; const float* b2_tc2_b = (const float*)d_in[25];
    const float* fc1_w = (const float*)d_in[26];    const float* fc1_b = (const float*)d_in[27];
    const float* hbn1_g = (const float*)d_in[28];   const float* hbn1_b = (const float*)d_in[29];
    const float* fc2_w = (const float*)d_in[30];    const float* fc2_b = (const float*)d_in[31];
    const float* hbn2_g = (const float*)d_in[32];   const float* hbn2_b = (const float*)d_in[33];
    const float* fc3_w = (const float*)d_in[34];    const float* fc3_b = (const float*)d_in[35];
    float* out = (float*)d_out;

    // ---- workspace partition (peak ~140 MB, L2-resident working set) ----
    char* ws = (char*)d_ws;
    constexpr size_t SZ_BIG = (size_t)NND * 64 * sizeof(float);          // 29.5 MB
    constexpr size_t OFF_BUF0 = 0;
    constexpr size_t OFF_BUF1 = OFF_BUF0 + SZ_BIG;
    constexpr size_t OFF_XL   = OFF_BUF1 + SZ_BIG;
    constexpr size_t OFF_XR   = OFF_XL   + SZ_BIG;
    constexpr size_t OFF_H16  = OFF_XR   + SZ_BIG;                       // N*64 f16
    constexpr size_t OFF_EX   = OFF_H16  + (size_t)NND * 64 * 2;
    constexpr size_t OFF_SSUM = OFF_EX   + (size_t)ENTOT * 4;
    constexpr size_t OFF_STAT = OFF_SSUM + (size_t)NND * 4;
    constexpr size_t OFF_W16A = OFF_STAT + 512;
    constexpr size_t OFF_W16B = OFF_W16A + 8192;
    constexpr size_t OFF_H1   = OFF_W16B + 8192;                         // 128*128 f32
    constexpr size_t OFF_H1H  = OFF_H1   + 65536;                        // 128*128 f16
    constexpr size_t OFF_W2H  = OFF_H1H  + 32768;                        // 64*128 f16
    constexpr size_t OFF_H2   = OFF_W2H  + 16384;                        // 128*64 f32
    constexpr size_t OFF_W16C = OFF_H2   + 32768;                        // conv w f16 (<=40KB)

    float*    BUF0 = (float*)(ws + OFF_BUF0);
    float*    BUF1 = (float*)(ws + OFF_BUF1);
    float*    XL   = (float*)(ws + OFF_XL);
    float*    XR   = (float*)(ws + OFF_XR);
    _Float16* H16  = (_Float16*)(ws + OFF_H16);
    float*    EX   = (float*)(ws + OFF_EX);
    float*    SSUM = (float*)(ws + OFF_SSUM);
    float*    STAT = (float*)(ws + OFF_STAT);
    _Float16* W16A = (_Float16*)(ws + OFF_W16A);
    _Float16* W16B = (_Float16*)(ws + OFF_W16B);
    float*    H1   = (float*)(ws + OFF_H1);
    _Float16* H1H  = (_Float16*)(ws + OFF_H1H);
    _Float16* W2H  = (_Float16*)(ws + OFF_W2H);
    float*    H2   = (float*)(ws + OFF_H2);
    _Float16* W16C = (_Float16*)(ws + OFF_W16C);
    _Float16* IM   = (_Float16*)(ws + OFF_XL);  // im2col: reuses XL+XR(+H16) when dead
    _Float16* X16  = H16;                       // FC stage reuse
    _Float16* FCW16 = (_Float16*)(ws + OFF_XL); // FC stage reuse

    const unsigned TH = 256;

    // ================= block 1 (C = 32) =================
    conv1d_relu_k<3, 32><<<gblk((size_t)BB * 32 * TT, TH), TH, 0, stream>>>(x, b1_tc1_w, b1_tc1_b, BUF0);
    f32_to_f16_k<<<gblk((size_t)NND * 32, TH), TH, 0, stream>>>(BUF0, H16, (size_t)NND * 32);
    f32_to_f16_k<<<gblk(32 * 32, TH), TH, 0, stream>>>(b1_wl, W16A, 32 * 32);
    f32_to_f16_k<<<gblk(32 * 32, TH), TH, 0, stream>>>(b1_wr, W16B, 32 * 32);
    gat_linear_wmma_k<32><<<dim3(NND / 16, 2), 32, 0, stream>>>(H16, W16A, b1_bl, W16B, b1_br, XL, XR);
    zero_f32_k<<<gblk((size_t)NND, TH), TH, 0, stream>>>(SSUM, (size_t)NND);
    zero_f32_k<<<gblk((size_t)NND * 32, TH), TH, 0, stream>>>(BUF1, (size_t)NND * 32);
    zero_f32_k<<<1, TH, 0, stream>>>(STAT, 64);
    edge_scores_k<32><<<gblk(ENTOT, TH), TH, 0, stream>>>(eidx, XL, XR, b1_att, EX, SSUM);
    edge_aggregate_k<32><<<gblk(ENTOT, TH), TH, 0, stream>>>(eidx, XL, EX, SSUM, BUF1);
    gat_post_stats_k<32><<<gblk((size_t)NND * 32, TH), TH, 0, stream>>>(BUF1, b1_gbias, STAT);
    bn_apply_k<32><<<gblk((size_t)NND * 32, TH), TH, 0, stream>>>(BUF1, STAT, b1_bn_g, b1_bn_b);
    // tc2 (32->32, K=160) as WMMA GEMM, async-LDS staged A tiles
    im2col_f16_k<32><<<gblk((size_t)NND * 160, TH), TH, 0, stream>>>(BUF1, IM);
    f32_to_f16_k<<<gblk(32 * 160, TH), TH, 0, stream>>>(b1_tc2_w, W16C, 32 * 160);
    conv_wmma_k<160, 32><<<dim3(NND / 16, 2), 32, 0, stream>>>(IM, W16C, b1_tc2_b, BUF0);

    // ================= block 2 (C = 64) =================
    // tc1 (32->64, K=160)
    im2col_f16_k<32><<<gblk((size_t)NND * 160, TH), TH, 0, stream>>>(BUF0, IM);
    f32_to_f16_k<<<gblk(64 * 160, TH), TH, 0, stream>>>(b2_tc1_w, W16C, 64 * 160);
    conv_wmma_k<160, 64><<<dim3(NND / 16, 4), 32, 0, stream>>>(IM, W16C, b2_tc1_b, BUF1);
    f32_to_f16_k<<<gblk((size_t)NND * 64, TH), TH, 0, stream>>>(BUF1, H16, (size_t)NND * 64);
    f32_to_f16_k<<<gblk(64 * 64, TH), TH, 0, stream>>>(b2_wl, W16A, 64 * 64);
    f32_to_f16_k<<<gblk(64 * 64, TH), TH, 0, stream>>>(b2_wr, W16B, 64 * 64);
    gat_linear_wmma_k<64><<<dim3(NND / 16, 4), 32, 0, stream>>>(H16, W16A, b2_bl, W16B, b2_br, XL, XR);
    zero_f32_k<<<gblk((size_t)NND, TH), TH, 0, stream>>>(SSUM, (size_t)NND);
    zero_f32_k<<<gblk((size_t)NND * 64, TH), TH, 0, stream>>>(BUF0, (size_t)NND * 64);
    zero_f32_k<<<1, TH, 0, stream>>>(STAT, 128);
    edge_scores_k<64><<<gblk(ENTOT, TH), TH, 0, stream>>>(eidx, XL, XR, b2_att, EX, SSUM);
    edge_aggregate_k<64><<<gblk(ENTOT, TH), TH, 0, stream>>>(eidx, XL, EX, SSUM, BUF0);
    gat_post_stats_k<64><<<gblk((size_t)NND * 64, TH), TH, 0, stream>>>(BUF0, b2_gbias, STAT);
    bn_apply_k<64><<<gblk((size_t)NND * 64, TH), TH, 0, stream>>>(BUF0, STAT, b2_bn_g, b2_bn_b);
    // tc2 (64->64, K=320)
    im2col_f16_k<64><<<gblk((size_t)NND * 320, TH), TH, 0, stream>>>(BUF0, IM);
    f32_to_f16_k<<<gblk(64 * 320, TH), TH, 0, stream>>>(b2_tc2_w, W16C, 64 * 320);
    conv_wmma_k<320, 64><<<dim3(NND / 16, 4), 32, 0, stream>>>(IM, W16C, b2_tc2_b, BUF1);

    // ================= FC head =================
    f32_to_f16_k<<<gblk((size_t)BB * 57600, TH), TH, 0, stream>>>(BUF1, X16, (size_t)BB * 57600);
    f32_to_f16_k<<<gblk((size_t)128 * 57600, TH), TH, 0, stream>>>(fc1_w, FCW16, (size_t)128 * 57600);
    zero_f32_k<<<gblk(128 * 128, TH), TH, 0, stream>>>(H1, 128 * 128);
    fc1_wmma_splitk_k<<<dim3(8, 8, 8), 32, 0, stream>>>(X16, FCW16, H1);
    bias_relu_k<<<gblk(128 * 128, TH), TH, 0, stream>>>(H1, fc1_b, 128, 128);
    head_bn_k<<<1, 128, 0, stream>>>(H1, hbn1_g, hbn1_b, 128, 128);
    f32_to_f16_k<<<gblk(128 * 128, TH), TH, 0, stream>>>(H1, H1H, 128 * 128);
    f32_to_f16_k<<<gblk(64 * 128, TH), TH, 0, stream>>>(fc2_w, W2H, 64 * 128);
    fc2_wmma_k<<<dim3(8, 4), 32, 0, stream>>>(H1H, W2H, fc2_b, H2);
    head_bn_k<<<1, 64, 0, stream>>>(H2, hbn2_g, hbn2_b, 128, 64);
    fc3_out_k<<<gblk(BB * 3, TH), TH, 0, stream>>>(H2, fc3_w, fc3_b, out);
}
